// Model_39676907882138
// MI455X (gfx1250) — matrix operand order repaired and checked by reading, not compiled, not run on hardware
//
#include <hip/hip_runtime.h>
#include <stdint.h>

#define S_LEN 4096
#define DHEAD 64
#define NHEAD 16
#define KTILE 64
#define QROWS_PER_BLOCK 128            // 8 waves * 16 query rows
#define NITER (S_LEN / KTILE)          // 64
#define KROW 72                        // padded Ks row stride (f16 elems): 144B, 16B-aligned
#define VROW 72                        // padded VTs row stride (f16 elems): 144B, 16B-aligned
// 1/(sqrt(64)*2) * log2(e): softmax computed in base-2 domain
#define QSCALE (0.0625f * 1.44269504088896340736f)

typedef __attribute__((ext_vector_type(16))) _Float16 v16h;
typedef __attribute__((ext_vector_type(8)))  float    f32x8;

union FragA {
  v16h     v;
  uint32_t u[8];
};

// single v_cvt_pk_rtz_f16_f32
__device__ __forceinline__ uint32_t pack_f16(float lo, float hi) {
  return __builtin_bit_cast(uint32_t, __builtin_amdgcn_cvt_pkrtz(lo, hi));
}
// exchange with the other 16-lane half (identity permutation) — pure VALU
__device__ __forceinline__ uint32_t xhalf_u32(uint32_t x) {
  return (uint32_t)__builtin_amdgcn_permlanex16((int)x, (int)x, 0x76543210, 0xFEDCBA98, false, false);
}
__device__ __forceinline__ float xhalf_f32(float x) {
  int i = __builtin_bit_cast(int, x);
  int j = __builtin_amdgcn_permlanex16(i, i, 0x76543210, 0xFEDCBA98, false, false);
  return __builtin_bit_cast(float, j);
}

__global__ __launch_bounds__(256) void fa_fwd_kernel(const float* __restrict__ q,
                                                     const float* __restrict__ k,
                                                     const float* __restrict__ v,
                                                     float* __restrict__ out) {
  // Double-buffered LDS: K tile row-major f16 [key][d], V tile transposed f16 [d][key]
  __shared__ __align__(16) unsigned short Ks[2][KTILE * KROW];   // 2 x 9 KB
  __shared__ __align__(16) unsigned short VTs[2][DHEAD * VROW];  // 2 x 9 KB

  const int tid  = threadIdx.x;
  const int wid  = tid >> 5;
  const int lane = tid & 31;
  const int ln   = lane & 15;   // qrow col (B/C layout N) and A-layout row M
  const int hs   = lane >> 4;   // half-wave selector

  const int h  = blockIdx.y;
  const int q0 = blockIdx.x * QROWS_PER_BLOCK + wid * 16;

  const float* kh = k + (size_t)h * S_LEN * DHEAD;
  const float* vh = v + (size_t)h * S_LEN * DHEAD;

  // per-thread staging coordinates (constant across iterations)
  int k_key[4], k_dq[4];
#pragma unroll
  for (int i = 0; i < 4; ++i) { int idx4 = i * 256 + tid; k_key[i] = idx4 >> 4; k_dq[i] = (idx4 & 15) * 4; }
  int v_d[8], v_k2[8];
#pragma unroll
  for (int i = 0; i < 8; ++i) { int idx2 = i * 256 + tid; v_d[i] = idx2 & 63; v_k2[i] = (idx2 >> 6) * 2; }

  // ---- Q^T as WMMA B operand: lane = qrow, element e of step s <-> d = s*32 + hs*16 + e ----
  FragA qb0, qb1;
  {
    const float* qrow = q + ((size_t)h * S_LEN + q0 + ln) * DHEAD;
#pragma unroll
    for (int i = 0; i < 8; ++i) {
      qb0.u[i] = pack_f16(qrow[hs * 16 + 2 * i] * QSCALE,      qrow[hs * 16 + 2 * i + 1] * QSCALE);
      qb1.u[i] = pack_f16(qrow[32 + hs * 16 + 2 * i] * QSCALE, qrow[32 + hs * 16 + 2 * i + 1] * QSCALE);
    }
  }

  // ---- prologue: stage tile 0 into buffer 0 ----
  {
#pragma unroll
    for (int i = 0; i < 4; ++i) {
      float4 f = *(const float4*)(kh + k_key[i] * DHEAD + k_dq[i]);
      uint32_t* d = (uint32_t*)&Ks[0][k_key[i] * KROW + k_dq[i]];
      d[0] = pack_f16(f.x, f.y); d[1] = pack_f16(f.z, f.w);
    }
#pragma unroll
    for (int i = 0; i < 8; ++i) {
      float a = vh[v_k2[i] * DHEAD + v_d[i]];
      float b = vh[(v_k2[i] + 1) * DHEAD + v_d[i]];
      *(uint32_t*)&VTs[0][v_d[i] * VROW + v_k2[i]] = pack_f16(a, b);
    }
  }

  // O^T accumulators: tile t covers d = t*16 + (r + hs*8), col = qrow = ln
  f32x8 O0 = {}, O1 = {}, O2 = {}, O3 = {};
  float mrun = -3.0e38f, lrun = 0.0f;   // per-lane (per qrow) stats, base-2 domain

  for (int kt = 0; kt < NITER; ++kt) {
    const int buf = kt & 1;
    __syncthreads();   // buffer `buf` ready; buffer buf^1 free to overwrite

    // ---- issue next tile's global loads early (latency hides under compute) ----
    const bool hasNext = (kt + 1) < NITER;
    float4 nf[4]; float nva[8], nvb[8];
    if (hasNext) {
      const float* kt_g = kh + (size_t)(kt + 1) * (KTILE * DHEAD);
      const float* vt_g = vh + (size_t)(kt + 1) * (KTILE * DHEAD);
#pragma unroll
      for (int i = 0; i < 4; ++i) nf[i] = *(const float4*)(kt_g + k_key[i] * DHEAD + k_dq[i]);
#pragma unroll
      for (int i = 0; i < 8; ++i) {
        nva[i] = vt_g[v_k2[i] * DHEAD + v_d[i]];
        nvb[i] = vt_g[(v_k2[i] + 1) * DHEAD + v_d[i]];
      }
      if (kt + 2 < NITER) {   // prefetch two tiles ahead, near locality
        __builtin_prefetch(kt_g + KTILE * DHEAD + tid * 16, 0, 3);
        __builtin_prefetch(vt_g + KTILE * DHEAD + tid * 16, 0, 3);
      }
    }

    // ---- S^T = K_tile . Q^T : four 16(key)x16(qrow) tiles over 64 keys ----
    f32x8 st0, st1, st2, st3;
    {
      f32x8 zz = {};
#pragma unroll
      for (int n = 0; n < 4; ++n) {
        const uint32_t* rn = (const uint32_t*)&Ks[buf][(n * 16 + ln) * KROW];
        FragA ka0, ka1;
#pragma unroll
        for (int i = 0; i < 4; ++i) {
          ka0.u[i] = rn[hs * 8 + i];      ka0.u[4 + i] = rn[hs * 8 + 4 + i];       // d = hs*16 + ...
          ka1.u[i] = rn[16 + hs * 8 + i]; ka1.u[4 + i] = rn[16 + hs * 8 + 4 + i];  // d += 32
        }
        f32x8 s = __builtin_amdgcn_wmma_f32_16x16x32_f16(false, ka0.v, false, qb0.v, (short)0, zz, false, false);
        s       = __builtin_amdgcn_wmma_f32_16x16x32_f16(false, ka1.v, false, qb1.v, (short)0, s,  false, false);
        if (n == 0) st0 = s; else if (n == 1) st1 = s; else if (n == 2) st2 = s; else st3 = s;
      }
    }

    // ---- online softmax (base-2). st_n[r]: key = n*16 + r + hs*8, qrow = ln ----
    float tm = fmaxf(fmaxf(st0[0], st1[0]), fmaxf(st2[0], st3[0]));
#pragma unroll
    for (int r = 1; r < 8; ++r)
      tm = fmaxf(tm, fmaxf(fmaxf(st0[r], st1[r]), fmaxf(st2[r], st3[r])));
    tm = fmaxf(tm, xhalf_f32(tm));           // combine the two key-halves
    float mnew  = fmaxf(mrun, tm);
    float alpha = __builtin_amdgcn_exp2f(mrun - mnew);

    // ---- exp + pack P^T chunks (B operand: lane = qrow, element e: key = c*32 + hs*16 + e) ----
    FragA pb0, pb1;
    float ls = 0.0f;
#pragma unroll
    for (int i = 0; i < 4; ++i) {            // chunk 0: tiles 0,1 (keys 0..31)
      float a0 = __builtin_amdgcn_exp2f(st0[2 * i]     - mnew);
      float a1 = __builtin_amdgcn_exp2f(st0[2 * i + 1] - mnew);
      float b0 = __builtin_amdgcn_exp2f(st1[2 * i]     - mnew);
      float b1 = __builtin_amdgcn_exp2f(st1[2 * i + 1] - mnew);
      ls += (a0 + a1) + (b0 + b1);
      uint32_t A = pack_f16(a0, a1);         // keys hs*8 + 2i, +1
      uint32_t B = pack_f16(b0, b1);         // keys 16 + hs*8 + 2i, +1
      uint32_t Z = hs ? A : B;               // what the partner half needs
      uint32_t cZ = xhalf_u32(Z);
      pb0.u[i]     = hs ? cZ : A;
      pb0.u[4 + i] = hs ? B  : cZ;
    }
#pragma unroll
    for (int i = 0; i < 4; ++i) {            // chunk 1: tiles 2,3 (keys 32..63)
      float a0 = __builtin_amdgcn_exp2f(st2[2 * i]     - mnew);
      float a1 = __builtin_amdgcn_exp2f(st2[2 * i + 1] - mnew);
      float b0 = __builtin_amdgcn_exp2f(st3[2 * i]     - mnew);
      float b1 = __builtin_amdgcn_exp2f(st3[2 * i + 1] - mnew);
      ls += (a0 + a1) + (b0 + b1);
      uint32_t A = pack_f16(a0, a1);
      uint32_t B = pack_f16(b0, b1);
      uint32_t Z = hs ? A : B;
      uint32_t cZ = xhalf_u32(Z);
      pb1.u[i]     = hs ? cZ : A;
      pb1.u[4 + i] = hs ? B  : cZ;
    }
    ls += xhalf_f32(ls);
    lrun = lrun * alpha + ls;
    mrun = mnew;
    O0 *= alpha; O1 *= alpha; O2 *= alpha; O3 *= alpha;

    // ---- O^T += V^T . P^T : A = V^T tile t (16 d x 32 keys), two key-chunks ----
#pragma unroll
    for (int t = 0; t < 4; ++t) {
      const uint32_t* vp = (const uint32_t*)&VTs[buf][(t * 16 + ln) * VROW];
      FragA va, vb;
#pragma unroll
      for (int i = 0; i < 4; ++i) {
        va.u[i] = vp[hs * 8 + i];      va.u[4 + i] = vp[hs * 8 + 4 + i];       // keys 0..31
        vb.u[i] = vp[16 + hs * 8 + i]; vb.u[4 + i] = vp[16 + hs * 8 + 4 + i];  // keys 32..63
      }
      f32x8 acc = (t == 0) ? O0 : (t == 1) ? O1 : (t == 2) ? O2 : O3;
      acc = __builtin_amdgcn_wmma_f32_16x16x32_f16(false, va.v, false, pb0.v, (short)0, acc, false, false);
      acc = __builtin_amdgcn_wmma_f32_16x16x32_f16(false, vb.v, false, pb1.v, (short)0, acc, false, false);
      if (t == 0) O0 = acc; else if (t == 1) O1 = acc; else if (t == 2) O2 = acc; else O3 = acc;
    }

    // ---- convert+store next tile into the other buffer (consumed after next barrier) ----
    if (hasNext) {
      const int nbuf = buf ^ 1;
#pragma unroll
      for (int i = 0; i < 4; ++i) {
        uint32_t* d = (uint32_t*)&Ks[nbuf][k_key[i] * KROW + k_dq[i]];
        d[0] = pack_f16(nf[i].x, nf[i].y); d[1] = pack_f16(nf[i].z, nf[i].w);
      }
#pragma unroll
      for (int i = 0; i < 8; ++i)
        *(uint32_t*)&VTs[nbuf][v_d[i] * VROW + v_k2[i]] = pack_f16(nva[i], nvb[i]);
    }
  }

  // ---- epilogue: O^T element (d = t*16 + hs*8 + r, qrow = ln); contiguous b128 stores ----
  {
    float il = 1.0f / lrun;
    float* orow = out + ((size_t)h * S_LEN + q0 + ln) * DHEAD + hs * 8;
    union { f32x8 v; float4 f4[2]; } o;
    o.v = O0 * il; *(float4*)(orow + 0)  = o.f4[0]; *(float4*)(orow + 4)  = o.f4[1];
    o.v = O1 * il; *(float4*)(orow + 16) = o.f4[0]; *(float4*)(orow + 20) = o.f4[1];
    o.v = O2 * il; *(float4*)(orow + 32) = o.f4[0]; *(float4*)(orow + 36) = o.f4[1];
    o.v = O3 * il; *(float4*)(orow + 48) = o.f4[0]; *(float4*)(orow + 52) = o.f4[1];
  }
}

extern "C" void kernel_launch(void* const* d_in, const int* in_sizes, int n_in,
                              void* d_out, int out_size, void* d_ws, size_t ws_size,
                              hipStream_t stream) {
  (void)in_sizes; (void)n_in; (void)out_size; (void)d_ws; (void)ws_size;
  const float* q = (const float*)d_in[0];
  const float* k = (const float*)d_in[1];
  const float* v = (const float*)d_in[2];
  float* out = (float*)d_out;
  dim3 grid(S_LEN / QROWS_PER_BLOCK, NHEAD, 1);   // 32 x 16 blocks
  fa_fwd_kernel<<<grid, 256, 0, stream>>>(q, k, v, out);
}
